// SwitchHead_47974784697233
// MI455X (gfx1250) — matrix operand order repaired and checked by reading, not compiled, not run on hardware
//
#include <hip/hip_runtime.h>

typedef __bf16 bf16;
typedef __attribute__((ext_vector_type(16))) __bf16 v16bf;
typedef __attribute__((ext_vector_type(8)))  __bf16 v8bf;
typedef __attribute__((ext_vector_type(8)))  float  v8f;

#define NT     8192   // B*S tokens
#define DMODEL 1024
#define SEQ    2048
#define NB     4
#define NH     8
#define NE     8
#define DHEAD  128

// ---------- helpers ----------
__device__ __forceinline__ v8f vzero(){
  v8f z;
#pragma unroll
  for (int i=0;i<8;++i) z[i]=0.f;
  return z;
}

// Build a 16-half WMMA fragment from two 16B-aligned 8-half LDS loads.
__device__ __forceinline__ v16bf ld16(const bf16* p0, const bf16* p1){
  const v8bf a = *(const v8bf*)p0;
  const v8bf b = *(const v8bf*)p1;
  v16bf r;
#pragma unroll
  for (int i=0;i<8;++i){ r[i]=a[i]; r[i+8]=b[i]; }
  return r;
}

__device__ __forceinline__ v8f wmma_bf16(v16bf a, v16bf b, v8f c){
  return __builtin_amdgcn_wmma_f32_16x16x32_bf16(false, a, false, b, (short)0, c, false, false);
}

// ---------- f32 -> bf16 convert ----------
__global__ void k_cvt_bf16(const float* __restrict__ src, bf16* __restrict__ dst, int n4){
  int i = blockIdx.x*blockDim.x + threadIdx.x;
  if (i < n4){
    float4 v = ((const float4*)src)[i];
    union { bf16 h[4]; uint2 u; } r;
    r.h[0]=(bf16)v.x; r.h[1]=(bf16)v.y; r.h[2]=(bf16)v.z; r.h[3]=(bf16)v.w;
    ((uint2*)dst)[i] = r.u;
  }
}

// ---------- gates: sigmoid selectors + top-2 scatter ----------
__global__ __launch_bounds__(128) void k_gates(const float* __restrict__ x,
                                               const float* __restrict__ selv,
                                               const float* __restrict__ selo,
                                               float* __restrict__ gvf,
                                               float* __restrict__ gof){
  __shared__ float xs[DMODEL];
  __shared__ float g[128];
  const int t = blockIdx.x;
  const int tid = threadIdx.x;
  const float4* xr = (const float4*)(x + (size_t)t*DMODEL);
#pragma unroll
  for (int i=0;i<2;++i) ((float4*)xs)[tid + i*128] = xr[tid + i*128];
  __syncthreads();
  const float* wr = (tid < 64) ? (selv + (size_t)tid*DMODEL) : (selo + (size_t)(tid-64)*DMODEL);
  float acc = 0.f;
  for (int d=0; d<DMODEL; d+=4){
    acc += xs[d]*wr[d] + xs[d+1]*wr[d+1] + xs[d+2]*wr[d+2] + xs[d+3]*wr[d+3];
  }
  g[tid] = 1.f/(1.f + __expf(-acc));
  __syncthreads();
  if (tid < 16){
    const int h = tid & 7;
    const float* gg = &g[(tid < 8 ? 0 : 64) + h*NE];
    float* outp = (tid < 8 ? gvf : gof) + (size_t)t*(NH*NE) + h*NE;
    int i1 = 0; float b1 = gg[0];
#pragma unroll
    for (int e=1;e<NE;++e){ if (gg[e] > b1){ b1 = gg[e]; i1 = e; } }
    int i2 = (i1==0)?1:0; float b2 = gg[i2];
#pragma unroll
    for (int e=0;e<NE;++e){ if (e!=i1 && gg[e] > b2){ b2 = gg[e]; i2 = e; } }
#pragma unroll
    for (int e=0;e<NE;++e) outp[e] = (e==i1) ? b1 : ((e==i2) ? b2 : 0.f);
  }
}

// ---------- QK projection GEMM + RoPE (software-pipelined) ----------
__global__ __launch_bounds__(256) void k_qkproj(const bf16* __restrict__ xb,
                                                const bf16* __restrict__ wqb,
                                                const bf16* __restrict__ wkb,
                                                const int* __restrict__ positions,
                                                bf16* __restrict__ qb, bf16* __restrict__ kb){
  __shared__ bf16 As[128*40];
  __shared__ bf16 Bs[128*40];
  const int tid  = threadIdx.x;
  const int w    = tid >> 5, lane = tid & 31;
  const int l16  = lane & 15, hi = lane >> 4;
  const int m0   = blockIdx.x * 128;
  const int nt   = blockIdx.y;
  const bool isQ = nt < 8;
  const int h    = nt & 7;
  const bf16* W  = isQ ? wqb : wkb;
  const int n0   = h * 128;

  v8f acc[8];
#pragma unroll
  for (int j=0;j<8;++j) acc[j] = vzero();

  const int cm = tid >> 1, half = tid & 1;
  const int akb = hi ? 16 : 0, bkb = hi ? 16 : 0;

  const bf16* aSrc = xb + (size_t)(m0 + cm)*DMODEL + half*16;
  const bf16* bSrc = W  + (size_t)(n0 + cm)*DMODEL + half*16;
  uint4* dA = (uint4*)(&As[cm*40 + half*16]);
  uint4* dB = (uint4*)(&Bs[cm*40 + half*16]);

  // prologue: tile 0 into registers
  uint4 ra0 = ((const uint4*)aSrc)[0], ra1 = ((const uint4*)aSrc)[1];
  uint4 rb0 = ((const uint4*)bSrc)[0], rb1 = ((const uint4*)bSrc)[1];

  for (int k0 = 0; k0 < DMODEL; k0 += 32){
    __syncthreads();
    dA[0] = ra0; dA[1] = ra1;
    dB[0] = rb0; dB[1] = rb1;
    __syncthreads();
    if (k0 + 32 < DMODEL){       // fetch next tile while computing this one
      const bf16* aN = aSrc + k0 + 32;
      const bf16* bN = bSrc + k0 + 32;
      ra0 = ((const uint4*)aN)[0]; ra1 = ((const uint4*)aN)[1];
      rb0 = ((const uint4*)bN)[0]; rb1 = ((const uint4*)bN)[1];
      __builtin_prefetch(aN + 32, 0, 3);
    }
    v16bf a = ld16(&As[(w*16 + l16)*40 + akb], &As[(w*16 + l16)*40 + akb + 8]);
    v16bf bf_[8];
#pragma unroll
    for (int j=0;j<8;++j)
      bf_[j] = ld16(&Bs[(j*16 + l16)*40 + bkb], &Bs[(j*16 + l16)*40 + bkb + 8]);
#pragma unroll
    for (int j=0;j<8;++j) acc[j] = wmma_bf16(a, bf_[j], acc[j]);
  }
  // RoPE epilogue: pair (dh, dh+64) = tiles (j, j+4), same lane/row.
  bf16* dstbase = isQ ? qb : kb;
#pragma unroll
  for (int j=0;j<4;++j){
    const int i = j*16 + l16;                 // freq index 0..63 == dh
    const float invf = __expf(-(float)i * 0.14391157f);   // ln(10000)/64
#pragma unroll
    for (int r=0;r<8;++r){
      const int tg = m0 + w*16 + r + hi*8;
      const float pos = (float)positions[tg];
      float sn, cs;
      __sincosf(pos * invf, &sn, &cs);
      const float c1 = acc[j][r], c2 = acc[j+4][r];
      const int bidx = tg >> 11, sidx = tg & (SEQ-1);
      bf16* dst = dstbase + ((size_t)(bidx*NH + h)*SEQ + sidx)*DHEAD;
      dst[i]      = (bf16)(c1*cs - c2*sn);
      dst[i + 64] = (bf16)(c2*cs + c1*sn);
    }
  }
}

// ---------- V projection: all experts dense, top-2 combine in-register ----------
__global__ __launch_bounds__(256) void k_vproj(const bf16* __restrict__ xb,
                                               const bf16* __restrict__ wvb,
                                               const float* __restrict__ gvf,
                                               bf16* __restrict__ vb){
  __shared__ bf16 As[128*40];
  __shared__ bf16 Bs[128*40];
  __shared__ float sg[128*8];
  const int tid = threadIdx.x;
  const int w   = tid >> 5, lane = tid & 31;
  const int l16 = lane & 15, hi = lane >> 4;
  const int m0  = blockIdx.x * 128;
  const int h   = blockIdx.y;

#pragma unroll
  for (int i=0;i<4;++i){
    const int idx = tid*4 + i;
    sg[idx] = gvf[(size_t)(m0 + (idx>>3))*(NH*NE) + h*NE + (idx&7)];
  }

  v8f facc[8];
#pragma unroll
  for (int j=0;j<8;++j) facc[j] = vzero();

  const int cm = tid >> 1, half = tid & 1;
  const int tk = tid >> 3, tn = (tid & 7)*16;
  const int akb = hi ? 16 : 0, bkb = hi ? 16 : 0;

  const bf16* aSrc = xb + (size_t)(m0 + cm)*DMODEL + half*16;
  uint4* dA = (uint4*)(&As[cm*40 + half*16]);

  for (int e=0;e<NE;++e){
    const bf16* We = wvb + (size_t)(h*NE + e) * DMODEL * DHEAD;
    v8f acc[8];
#pragma unroll
    for (int j=0;j<8;++j) acc[j] = vzero();

    // prologue: tile 0 regs
    uint4 ra0 = ((const uint4*)aSrc)[0], ra1 = ((const uint4*)aSrc)[1];
    const v8bf* bS0 = (const v8bf*)(We + (size_t)tk*DHEAD + tn);
    v8bf ru0 = bS0[0], ru1 = bS0[1];

    for (int k0=0;k0<DMODEL;k0+=32){
      __syncthreads();
      dA[0] = ra0; dA[1] = ra1;
#pragma unroll
      for (int i=0;i<8;++i){ Bs[(tn+i)*40 + tk] = ru0[i]; Bs[(tn+8+i)*40 + tk] = ru1[i]; }
      __syncthreads();
      if (k0 + 32 < DMODEL){
        const bf16* aN = aSrc + k0 + 32;
        ra0 = ((const uint4*)aN)[0]; ra1 = ((const uint4*)aN)[1];
        const v8bf* bN = (const v8bf*)(We + (size_t)(k0 + 32 + tk)*DHEAD + tn);
        ru0 = bN[0]; ru1 = bN[1];
      }
      v16bf a = ld16(&As[(w*16+l16)*40 + akb], &As[(w*16+l16)*40 + akb + 8]);
      v16bf bf_[8];
#pragma unroll
      for (int j=0;j<8;++j)
        bf_[j] = ld16(&Bs[(j*16+l16)*40 + bkb], &Bs[(j*16+l16)*40 + bkb + 8]);
#pragma unroll
      for (int j=0;j<8;++j) acc[j] = wmma_bf16(a, bf_[j], acc[j]);
    }
#pragma unroll
    for (int r=0;r<8;++r){
      const float gf = sg[(w*16 + r + hi*8)*NE + e];
#pragma unroll
      for (int j=0;j<8;++j) facc[j][r] += gf * acc[j][r];
    }
  }
#pragma unroll
  for (int r=0;r<8;++r){
    const int tg = m0 + w*16 + r + hi*8;
    const int bidx = tg >> 11, sidx = tg & (SEQ-1);
    bf16* dst = vb + ((size_t)(bidx*NH + h)*SEQ + sidx)*DHEAD;
#pragma unroll
    for (int j=0;j<8;++j) dst[j*16 + l16] = (bf16)facc[j][r];
  }
}

// ---------- causal flash attention + gated ctxg scatter ----------
__global__ __launch_bounds__(128) void k_attn(const bf16* __restrict__ qb,
                                              const bf16* __restrict__ kbuf,
                                              const bf16* __restrict__ vbuf,
                                              const float* __restrict__ gof,
                                              bf16* __restrict__ ctxg){
  __shared__ bf16 Qs[64*128];
  __shared__ bf16 Ks[32*128];
  __shared__ bf16 Vs[128*40];
  __shared__ bf16 Ps[4][16*40];
  __shared__ float Go[64*8];
  const int tid = threadIdx.x;
  const int w = tid >> 5, lane = tid & 31, l16 = lane & 15, hi = lane >> 4;
  const int bh = blockIdx.y;
  const int bidx = bh >> 3, h = bh & 7;
  const int q0 = blockIdx.x * 64;
  const bf16* Qg = qb + ((size_t)bh*SEQ + q0)*DHEAD;
  { // stage Q tile [64][128]
    const int m = tid >> 1, seg = tid & 1;
    const uint4* s = (const uint4*)(Qg + (size_t)m*DHEAD + seg*64);
    uint4* d = (uint4*)(&Qs[m*128 + seg*64]);
#pragma unroll
    for (int i=0;i<8;++i) d[i] = s[i];
  }
#pragma unroll
  for (int i=0;i<4;++i){
    const int idx = tid*4 + i;
    Go[idx] = gof[(size_t)(bidx*SEQ + q0 + (idx>>3))*(NH*NE) + h*NE + (idx&7)];
  }
  v8f oacc[8];
#pragma unroll
  for (int j=0;j<8;++j) oacc[j] = vzero();
  float mrow[8], lrow[8];
#pragma unroll
  for (int r=0;r<8;++r){ mrow[r] = -1e38f; lrow[r] = 0.f; }

  const int nkt = (q0 + 64) / 32;
  const bf16* Kg = kbuf + (size_t)bh*SEQ*DHEAD;
  const bf16* Vg = vbuf + (size_t)bh*SEQ*DHEAD;
  const int akb = hi ? 16 : 0, bkb = hi ? 16 : 0;

  // staging thread mapping
  const int kn = tid >> 2, kseg = tid & 3;                 // K: row, 32-half segment
  const int vkk = tid >> 2, vd0 = (tid & 3)*32;            // V: key row, 32-d segment
  const bf16* kSrc = Kg + (size_t)kn*DHEAD + kseg*32;
  const bf16* vSrc = Vg + (size_t)vkk*DHEAD + vd0;

  // prologue: k-tile 0 into registers
  uint4 rk0 = ((const uint4*)kSrc)[0], rk1 = ((const uint4*)kSrc)[1],
        rk2 = ((const uint4*)kSrc)[2], rk3 = ((const uint4*)kSrc)[3];
  v8bf  rv0 = ((const v8bf*)vSrc)[0], rv1 = ((const v8bf*)vSrc)[1],
        rv2 = ((const v8bf*)vSrc)[2], rv3 = ((const v8bf*)vSrc)[3];

  for (int kt=0; kt<nkt; ++kt){
    const int k0 = kt*32;
    __syncthreads();
    { // commit staged K tile [32][128]
      uint4* d = (uint4*)(&Ks[kn*128 + kseg*32]);
      d[0]=rk0; d[1]=rk1; d[2]=rk2; d[3]=rk3;
    }
    { // commit staged V transposed: Vs[d][kk]
#pragma unroll
      for (int i=0;i<8;++i){
        Vs[(vd0 +      i)*40 + vkk] = rv0[i];
        Vs[(vd0 +  8 + i)*40 + vkk] = rv1[i];
        Vs[(vd0 + 16 + i)*40 + vkk] = rv2[i];
        Vs[(vd0 + 24 + i)*40 + vkk] = rv3[i];
      }
    }
    __syncthreads();
    if (kt + 1 < nkt){   // fetch next K/V tile behind the compute
      const bf16* kN = kSrc + (size_t)(k0 + 32)*DHEAD;
      const bf16* vN = vSrc + (size_t)(k0 + 32)*DHEAD;
      rk0=((const uint4*)kN)[0]; rk1=((const uint4*)kN)[1];
      rk2=((const uint4*)kN)[2]; rk3=((const uint4*)kN)[3];
      rv0=((const v8bf*)vN)[0];  rv1=((const v8bf*)vN)[1];
      rv2=((const v8bf*)vN)[2];  rv3=((const v8bf*)vN)[3];
    }
    v8f sacc[2]; sacc[0] = vzero(); sacc[1] = vzero();
#pragma unroll
    for (int dc=0;dc<4;++dc){
      const int d0 = dc*32;
      v16bf a  = ld16(&Qs[(w*16+l16)*128 + d0 + akb], &Qs[(w*16+l16)*128 + d0 + akb + 8]);
      v16bf b0 = ld16(&Ks[(     l16)*128 + d0 + bkb], &Ks[(     l16)*128 + d0 + bkb + 8]);
      v16bf b1 = ld16(&Ks[(16 + l16)*128 + d0 + bkb], &Ks[(16 + l16)*128 + d0 + bkb + 8]);
      sacc[0] = wmma_bf16(a, b0, sacc[0]);
      sacc[1] = wmma_bf16(a, b1, sacc[1]);
    }
    // scale + causal mask (select, no branch)
#pragma unroll
    for (int j=0;j<2;++j){
      const int ki = k0 + j*16 + l16;
#pragma unroll
      for (int r=0;r<8;++r){
        const int qi = q0 + w*16 + r + hi*8;
        const float sv = sacc[j][r] * 0.08838835f;
        sacc[j][r] = (ki > qi) ? -1e38f : sv;
      }
    }
    // online softmax
#pragma unroll
    for (int r=0;r<8;++r){
      float cm2 = fmaxf(sacc[0][r], sacc[1][r]);
      cm2 = fmaxf(cm2, __shfl_xor(cm2, 1, 32));
      cm2 = fmaxf(cm2, __shfl_xor(cm2, 2, 32));
      cm2 = fmaxf(cm2, __shfl_xor(cm2, 4, 32));
      cm2 = fmaxf(cm2, __shfl_xor(cm2, 8, 32));
      const float nm = fmaxf(mrow[r], cm2);
      const float alpha = __expf(mrow[r] - nm);
      mrow[r] = nm;
      float rs = 0.f;
#pragma unroll
      for (int j=0;j<2;++j){
        const float sv = sacc[j][r];
        const float p = (sv < -1e37f) ? 0.f : __expf(sv - nm);
        sacc[j][r] = p;
        rs += p;
      }
      rs += __shfl_xor(rs, 1, 32);
      rs += __shfl_xor(rs, 2, 32);
      rs += __shfl_xor(rs, 4, 32);
      rs += __shfl_xor(rs, 8, 32);
      lrow[r] = lrow[r]*alpha + rs;
#pragma unroll
      for (int j=0;j<8;++j) oacc[j][r] *= alpha;
    }
    // C-layout P -> A-layout via LDS bounce
#pragma unroll
    for (int j=0;j<2;++j)
#pragma unroll
      for (int r=0;r<8;++r)
        Ps[w][(r + hi*8)*40 + j*16 + l16] = (bf16)sacc[j][r];
    __syncthreads();
    { // O += P @ V  (batch fragment loads, then WMMAs)
      v16bf pa = ld16(&Ps[w][l16*40 + akb], &Ps[w][l16*40 + akb + 8]);
      v16bf vf[8];
#pragma unroll
      for (int j=0;j<8;++j)
        vf[j] = ld16(&Vs[(j*16+l16)*40 + bkb], &Vs[(j*16+l16)*40 + bkb + 8]);
#pragma unroll
      for (int j=0;j<8;++j) oacc[j] = wmma_bf16(pa, vf[j], oacc[j]);
    }
  }
  // epilogue: normalize + gated scatter into all E slots of ctxg
#pragma unroll
  for (int r=0;r<8;++r){
    const int m = w*16 + r + hi*8;
    const float rinv = 1.f / lrow[r];
    const size_t row = (size_t)(bidx*SEQ + q0 + m) * 8192;
#pragma unroll
    for (int j=0;j<8;++j){
      const float ov = oacc[j][r] * rinv;
#pragma unroll
      for (int e=0;e<NE;++e){
        ctxg[row + (size_t)((h*NE + e)*DHEAD) + j*16 + l16] = (bf16)(Go[m*NE + e] * ov);
      }
    }
  }
}

// ---------- O projection: [8192,8192] x [8192,1024] dense GEMM ----------
__global__ __launch_bounds__(256) void k_oproj(const bf16* __restrict__ ctxg,
                                               const bf16* __restrict__ wob,
                                               float* __restrict__ out){
  __shared__ bf16 As[128*40];
  __shared__ bf16 Bs[128*40];
  const int tid = threadIdx.x;
  const int w = tid >> 5, lane = tid & 31, l16 = lane & 15, hi = lane >> 4;
  const int m0 = blockIdx.x*128, n0 = blockIdx.y*128;
  v8f acc[8];
#pragma unroll
  for (int j=0;j<8;++j) acc[j] = vzero();
  const int cm = tid >> 1, half = tid & 1;
  const int tk = tid >> 3, tn = (tid & 7)*16;
  const int akb = hi ? 16 : 0, bkb = hi ? 16 : 0;

  const bf16* aSrc = ctxg + (size_t)(m0 + cm)*8192 + half*16;
  uint4* dA = (uint4*)(&As[cm*40 + half*16]);

  // prologue
  uint4 ra0 = ((const uint4*)aSrc)[0], ra1 = ((const uint4*)aSrc)[1];
  const v8bf* bS0 = (const v8bf*)(wob + (size_t)tk*DMODEL + n0 + tn);
  v8bf ru0 = bS0[0], ru1 = bS0[1];

  for (int k0=0;k0<8192;k0+=32){
    __syncthreads();
    dA[0] = ra0; dA[1] = ra1;
#pragma unroll
    for (int i=0;i<8;++i){ Bs[(tn+i)*40 + tk] = ru0[i]; Bs[(tn+8+i)*40 + tk] = ru1[i]; }
    __syncthreads();
    if (k0 + 32 < 8192){
      const bf16* aN = aSrc + k0 + 32;
      ra0 = ((const uint4*)aN)[0]; ra1 = ((const uint4*)aN)[1];
      const v8bf* bN = (const v8bf*)(wob + (size_t)(k0 + 32 + tk)*DMODEL + n0 + tn);
      ru0 = bN[0]; ru1 = bN[1];
      __builtin_prefetch(aN + 32, 0, 3);
    }
    v16bf a = ld16(&As[(w*16+l16)*40 + akb], &As[(w*16+l16)*40 + akb + 8]);
    v16bf bf_[8];
#pragma unroll
    for (int j=0;j<8;++j)
      bf_[j] = ld16(&Bs[(j*16+l16)*40 + bkb], &Bs[(j*16+l16)*40 + bkb + 8]);
#pragma unroll
    for (int j=0;j<8;++j) acc[j] = wmma_bf16(a, bf_[j], acc[j]);
  }
#pragma unroll
  for (int r=0;r<8;++r){
    float* dst = out + (size_t)(m0 + w*16 + r + hi*8)*DMODEL + n0;
#pragma unroll
    for (int j=0;j<8;++j) dst[j*16 + l16] = acc[j][r];
  }
}

// ---------- host launch ----------
extern "C" void kernel_launch(void* const* d_in, const int* in_sizes, int n_in,
                              void* d_out, int out_size, void* d_ws, size_t ws_size,
                              hipStream_t stream){
  const float* x  = (const float*)d_in[0];
  const float* Wq = (const float*)d_in[1];
  const float* Wk = (const float*)d_in[2];
  const float* Wv = (const float*)d_in[3];
  const float* Wo = (const float*)d_in[4];
  const float* sv = (const float*)d_in[5];
  const float* so = (const float*)d_in[6];
  const int*  pos = (const int*)d_in[7];
  float* out = (float*)d_out;
  (void)in_sizes; (void)n_in; (void)out_size; (void)ws_size;

  char* ws = (char*)d_ws;
  size_t off = 0;
  auto alloc = [&](size_t bytes)->char*{
    char* p = ws + off; off += (bytes + 255) & ~(size_t)255; return p;
  };
  bf16*  xb   = (bf16*) alloc((size_t)NT*DMODEL*2);
  bf16*  wqb  = (bf16*) alloc((size_t)1024*1024*2);
  bf16*  wkb  = (bf16*) alloc((size_t)1024*1024*2);
  bf16*  wvb  = (bf16*) alloc((size_t)64*1024*128*2);
  bf16*  wob  = (bf16*) alloc((size_t)64*128*1024*2);
  bf16*  qb   = (bf16*) alloc((size_t)NB*NH*SEQ*DHEAD*2);
  bf16*  kb   = (bf16*) alloc((size_t)NB*NH*SEQ*DHEAD*2);
  bf16*  vb   = (bf16*) alloc((size_t)NB*NH*SEQ*DHEAD*2);
  float* gvf  = (float*)alloc((size_t)NT*NH*NE*4);
  float* gof  = (float*)alloc((size_t)NT*NH*NE*4);
  bf16*  ctxg = (bf16*) alloc((size_t)NT*8192*2);

  auto cvt = [&](const float* s, bf16* dptr, size_t n){
    int n4 = (int)(n/4);
    k_cvt_bf16<<<(n4+255)/256, 256, 0, stream>>>(s, dptr, n4);
  };
  cvt(x,  xb,  (size_t)NT*DMODEL);
  cvt(Wq, wqb, (size_t)1024*1024);
  cvt(Wk, wkb, (size_t)1024*1024);
  cvt(Wv, wvb, (size_t)64*1024*128);
  cvt(Wo, wob, (size_t)64*128*1024);

  k_gates <<<NT, 128, 0, stream>>>(x, sv, so, gvf, gof);
  k_qkproj<<<dim3(64,16), 256, 0, stream>>>(xb, wqb, wkb, pos, qb, kb);
  k_vproj <<<dim3(64, 8), 256, 0, stream>>>(xb, wvb, gvf, vb);
  k_attn  <<<dim3(SEQ/64, NB*NH), 128, 0, stream>>>(qb, kb, vb, gof, ctxg);
  k_oproj <<<dim3(64, 8), 256, 0, stream>>>(ctxg, wob, out);
}
